// GraphHeterAttention_89919435309273
// MI455X (gfx1250) — matrix operand order repaired and checked by reading, not compiled, not run on hardware
//
#include <hip/hip_runtime.h>

#define Bq 32
#define Nn 2048
#define Ff 512
#define Ee 256

typedef unsigned short u16;
typedef unsigned int   u32;
typedef unsigned long long u64;
typedef __attribute__((ext_vector_type(16))) __bf16 v16bf;
typedef __attribute__((ext_vector_type(8)))  float  v8f;
typedef __attribute__((ext_vector_type(4)))  u32    v4u;
typedef __attribute__((ext_vector_type(8)))  int    v8i;
typedef __attribute__((ext_vector_type(4)))  int    v4i;

union FragBF {
    v16bf v;
    u16   u[16];
};
union Pack8 {
    uint4 q;
    u16   s[8];
};

__device__ __forceinline__ u16 bf16_rne(float f) {
    u32 u = __float_as_uint(f);
    u += 0x7FFFu + ((u >> 16) & 1u);
    return (u16)(u >> 16);
}
__device__ __forceinline__ float bf16_to_f(u16 h) {
    return __uint_as_float((u32)h << 16);
}
__device__ __forceinline__ v8f wmma_bf16(v16bf a, v16bf b, v8f c) {
    return __builtin_amdgcn_wmma_f32_16x16x32_bf16(false, a, false, b, (short)0, c, false, false);
}

// --- TDM: DMA a 2D tile of 256 rows x 32 u16 (row stride `stride` elems) into LDS.
// LDS padding: +4 DWORDs (8 u16) after every 16 DWORDs (32 u16) -> LDS row stride 40 u16,
// matching the compute-side addressing and avoiding bank conflicts.
__device__ __forceinline__ void tdm_load_tile(u32 lds_off, const u16* gptr, u32 stride_elems) {
    v4u g0;
    u64 ga = (u64)(size_t)gptr;
    g0.x = 1u;                                            // count=1 (valid user descriptor)
    g0.y = lds_off;                                       // lds_addr (bytes)
    g0.z = (u32)ga;                                       // global_addr[31:0]
    g0.w = ((u32)(ga >> 32) & 0x01FFFFFFu) | 0x80000000u; // global_addr[56:32] | type=2
    v8i g1;
    g1[0] = (int)((1u << 16) |        // data_size = 1 -> 2 bytes
                  (1u << 20) |        // pad_enable
                  (3u << 22) |        // pad_interval code 3 -> every 16 DWORDs
                  (3u << 25));        // pad_amount  code 3 -> 4 DWORDs
    g1[1] = 0;                        // atomic_barrier=0 ; tensor_dim0[15:0]=0
    g1[2] = 0x4000;                   // tensor_dim0 = 0x40000000 (no OOB), tensor_dim1[15:0]=0
    g1[3] = (32 << 16) | 0x4000;      // tensor_dim1 = 0x40000000 ; tile_dim0 = 32
    g1[4] = 256;                      // tile_dim1 = 256 ; tile_dim2 = 0
    g1[5] = (int)stride_elems;        // tensor_dim0_stride[31:0] (data_size units)
    g1[6] = 0;                        // stride0[47:32]=0 ; tensor_dim1_stride lo = 0
    g1[7] = 0;
    v4i g2;
    g2.x = 0x40000000; g2.y = 0; g2.z = 0; g2.w = 0;      // tensor_dim2 large, unused
    v4i g3;
    g3.x = 0; g3.y = 0; g3.z = 0; g3.w = 0;
    v8i g4;
    g4[0] = 0; g4[1] = 0; g4[2] = 0; g4[3] = 0;
    g4[4] = 0; g4[5] = 0; g4[6] = 0; g4[7] = 0;
    __builtin_amdgcn_tensor_load_to_lds(g0, g1, g2, g3, g4, 0);
}
__device__ __forceinline__ u32 lds_off_of(const void* p) {
    return (u32)(size_t)p;            // low 32 bits of generic addr = LDS byte offset
}

// ---------------- prep: W_w [F,E] fp32 -> Wt_hi/Wt_lo [E,F] bf16 ----------------
__global__ __launch_bounds__(256) void prepW_kernel(const float* __restrict__ W,
                                                    u16* __restrict__ Wt_hi,
                                                    u16* __restrict__ Wt_lo) {
    int t = blockIdx.x * 256 + threadIdx.x;   // 0 .. E*F-1
    int e = t >> 9;                           // / F
    int k = t & (Ff - 1);
    float f = W[(size_t)k * Ee + e];
    u16 h = bf16_rne(f);
    Wt_hi[t] = h;
    Wt_lo[t] = bf16_rne(f - bf16_to_f(h));
}

// ---------------- prep: adj [N,N] int -> maskT [i][j] bf16 (exact 0/1) ----------
__global__ __launch_bounds__(256) void prepM_kernel(const int* __restrict__ adj,
                                                    u16* __restrict__ maskT) {
    int t = blockIdx.x * 256 + threadIdx.x;   // 0 .. N*N-1
    int i = t >> 11;
    int j = t & (Nn - 1);
    maskT[t] = (adj[(size_t)j * Nn + i] > 0) ? (u16)0x3F80 : (u16)0;
}

// ---------------- GEMM1: h = x@W + b ; emits ht(bf16 hi/lo), score1, h_sum -------
// Wave tile: 16 rows x 128 cols (8 e-tiles). 8 waves: 4 row groups x 2 e-halves.
// Double-buffered TDM staging: DMA for chunk k+1 overlaps WMMA on chunk k.
__global__ __launch_bounds__(256) void gemm1_kernel(const float* __restrict__ x,      // [B*N,F]
                                                    const u16* __restrict__ Wt_hi,    // [E,F]
                                                    const u16* __restrict__ Wt_lo,
                                                    const float* __restrict__ Wb,     // [E]
                                                    const float* __restrict__ a1,     // [E,N]
                                                    u16* __restrict__ ht_hi,          // [B,E,N]
                                                    u16* __restrict__ ht_lo,
                                                    float* __restrict__ score,        // [B,N] atomic
                                                    float* __restrict__ h_sum) {      // [B,E] atomic
    __shared__ u16 lbuf[2][2][Ee * 40];       // [pingpong][hi/lo] : 80 KB
    const int tid   = threadIdx.x;
    const int lane  = tid & 31;
    const int wave  = tid >> 5;
    const int rw    = wave >> 1;
    const int ehalf = wave & 1;
    const int r0    = blockIdx.x * 64 + rw * 16;     // global row (b*N+n)
    const int b     = r0 >> 11;
    const int n0    = r0 & (Nn - 1);
    const int lrow  = lane & 15;
    const int half  = lane >> 4;

    v8f acc[8];
#pragma unroll
    for (int t = 0; t < 8; ++t) acc[t] = v8f{0.f,0.f,0.f,0.f,0.f,0.f,0.f,0.f};

    const float* xrow = x + (size_t)(r0 + lrow) * Ff;

    const int NC = Ff / 32;
    if (wave == 0) {                 // prologue: fill buffer 0
        tdm_load_tile(lds_off_of(lbuf[0][0]), Wt_hi, Ff);
        tdm_load_tile(lds_off_of(lbuf[0][1]), Wt_lo, Ff);
        __builtin_amdgcn_s_wait_tensorcnt(0);
    }
    __syncthreads();

    for (int kc = 0; kc < NC; ++kc) {
        const int cur = kc & 1;
        if (wave == 0 && kc + 1 < NC) {          // async-prefetch next chunk
            tdm_load_tile(lds_off_of(lbuf[cur ^ 1][0]), Wt_hi + (kc + 1) * 32, Ff);
            tdm_load_tile(lds_off_of(lbuf[cur ^ 1][1]), Wt_lo + (kc + 1) * 32, Ff);
        }
        const u16* lhi = &lbuf[cur][0][0];
        const u16* llo = &lbuf[cur][1][0];

        // A fragment: 16x32 fp32 chunk of x -> bf16 hi/lo split
        FragBF a_hi, a_lo;
        const int kb = kc * 32;
#pragma unroll
        for (int i = 0; i < 8; ++i) {
            float f0 = xrow[kb + half * 8 + i];
            u16 h0 = bf16_rne(f0);
            a_hi.u[i]     = h0;
            a_lo.u[i]     = bf16_rne(f0 - bf16_to_f(h0));
            float f1 = xrow[kb + 16 + half * 8 + i];
            u16 h1 = bf16_rne(f1);
            a_hi.u[8 + i] = h1;
            a_lo.u[8 + i] = bf16_rne(f1 - bf16_to_f(h1));
        }
#pragma unroll
        for (int et = 0; et < 8; ++et) {
            FragBF b_hi, b_lo;
            int e = (ehalf * 8 + et) * 16 + lrow;
            const uint4* ph = reinterpret_cast<const uint4*>(&lhi[e * 40 + half * 8]);
            const uint4* pl = reinterpret_cast<const uint4*>(&llo[e * 40 + half * 8]);
            *reinterpret_cast<uint4*>(&b_hi.u[0]) = ph[0];
            *reinterpret_cast<uint4*>(&b_hi.u[8]) = ph[2];
            *reinterpret_cast<uint4*>(&b_lo.u[0]) = pl[0];
            *reinterpret_cast<uint4*>(&b_lo.u[8]) = pl[2];
            acc[et] = wmma_bf16(a_hi.v, b_hi.v, acc[et]);
            acc[et] = wmma_bf16(a_hi.v, b_lo.v, acc[et]);
            acc[et] = wmma_bf16(a_lo.v, b_hi.v, acc[et]);
        }
        if (wave == 0) __builtin_amdgcn_s_wait_tensorcnt(0);
        __syncthreads();
    }

    // epilogue: +bias, write ht hi/lo (transposed), partial score1, h_sum
    float p[8];
#pragma unroll
    for (int m = 0; m < 8; ++m) p[m] = 0.f;

#pragma unroll
    for (int et = 0; et < 8; ++et) {
        int e = (ehalf * 8 + et) * 16 + lrow;
        float wb = Wb[e];
        const float* a1p = a1 + (size_t)e * Nn + n0 + half * 8;
        Pack8 ph, pl;
        float colsum = 0.f;
#pragma unroll
        for (int m = 0; m < 8; ++m) {
            float v = acc[et][m] + wb;
            u16 hb = bf16_rne(v);
            ph.s[m] = hb;
            pl.s[m] = bf16_rne(v - bf16_to_f(hb));
            colsum += v;
            p[m] += v * a1p[m];
        }
        size_t off = ((size_t)(b * Ee + e)) * Nn + n0 + half * 8;
        *reinterpret_cast<uint4*>(ht_hi + off) = ph.q;
        *reinterpret_cast<uint4*>(ht_lo + off) = pl.q;
        colsum += __shfl_xor(colsum, 16);
        if (half == 0) atomicAdd(&h_sum[b * Ee + e], colsum);
    }
#pragma unroll
    for (int m = 0; m < 8; ++m) {
        p[m] += __shfl_xor(p[m], 1);
        p[m] += __shfl_xor(p[m], 2);
        p[m] += __shfl_xor(p[m], 4);
        p[m] += __shfl_xor(p[m], 8);
    }
    if (lrow == 0) {
        int base = b * Nn + n0 + half * 8;
#pragma unroll
        for (int m = 0; m < 8; ++m) atomicAdd(&score[base + m], p[m]);
    }
}

// ---------------- GEMM2: score[b,i] += sum_e (maskT@h)[i,e] * a2[e,i] ------------
__global__ __launch_bounds__(256) void gemm2_kernel(const u16* __restrict__ maskT,   // [N,N]
                                                    const u16* __restrict__ ht_hi,   // [B,E,N]
                                                    const u16* __restrict__ ht_lo,
                                                    const float* __restrict__ a2,    // [E,N]
                                                    float* __restrict__ score) {     // [B,N] atomic
    __shared__ u16 lbuf[2][2][Ee * 40];       // [pingpong][hi/lo] : 80 KB
    const int tid   = threadIdx.x;
    const int lane  = tid & 31;
    const int wave  = tid >> 5;
    const int rw    = wave >> 1;
    const int ehalf = wave & 1;
    const int b     = blockIdx.x >> 5;
    const int i0    = (blockIdx.x & 31) * 64 + rw * 16;
    const int lrow  = lane & 15;
    const int half  = lane >> 4;

    v8f acc[8];
#pragma unroll
    for (int t = 0; t < 8; ++t) acc[t] = v8f{0.f,0.f,0.f,0.f,0.f,0.f,0.f,0.f};

    const u16* mrow = maskT + (size_t)(i0 + lrow) * Nn;
    const size_t hbase = (size_t)b * Ee * Nn;

    const int NC = Nn / 32;
    if (wave == 0) {
        tdm_load_tile(lds_off_of(lbuf[0][0]), ht_hi + hbase, Nn);
        tdm_load_tile(lds_off_of(lbuf[0][1]), ht_lo + hbase, Nn);
        __builtin_amdgcn_s_wait_tensorcnt(0);
    }
    __syncthreads();

    for (int jc = 0; jc < NC; ++jc) {
        const int cur = jc & 1;
        if (wave == 0 && jc + 1 < NC) {
            tdm_load_tile(lds_off_of(lbuf[cur ^ 1][0]), ht_hi + hbase + (jc + 1) * 32, Nn);
            tdm_load_tile(lds_off_of(lbuf[cur ^ 1][1]), ht_lo + hbase + (jc + 1) * 32, Nn);
        }
        const u16* lhi = &lbuf[cur][0][0];
        const u16* llo = &lbuf[cur][1][0];

        FragBF a_m;   // mask is exact in bf16: no split on A side
        *reinterpret_cast<uint4*>(&a_m.u[0]) =
            *reinterpret_cast<const uint4*>(mrow + jc * 32 + half * 8);
        *reinterpret_cast<uint4*>(&a_m.u[8]) =
            *reinterpret_cast<const uint4*>(mrow + jc * 32 + 16 + half * 8);
#pragma unroll
        for (int et = 0; et < 8; ++et) {
            FragBF b_hi, b_lo;
            int e = (ehalf * 8 + et) * 16 + lrow;
            const uint4* ph = reinterpret_cast<const uint4*>(&lhi[e * 40 + half * 8]);
            const uint4* pl = reinterpret_cast<const uint4*>(&llo[e * 40 + half * 8]);
            *reinterpret_cast<uint4*>(&b_hi.u[0]) = ph[0];
            *reinterpret_cast<uint4*>(&b_hi.u[8]) = ph[2];
            *reinterpret_cast<uint4*>(&b_lo.u[0]) = pl[0];
            *reinterpret_cast<uint4*>(&b_lo.u[8]) = pl[2];
            acc[et] = wmma_bf16(a_m.v, b_hi.v, acc[et]);
            acc[et] = wmma_bf16(a_m.v, b_lo.v, acc[et]);
        }
        if (wave == 0) __builtin_amdgcn_s_wait_tensorcnt(0);
        __syncthreads();
    }

    float p[8];
#pragma unroll
    for (int m = 0; m < 8; ++m) p[m] = 0.f;
#pragma unroll
    for (int et = 0; et < 8; ++et) {
        int e = (ehalf * 8 + et) * 16 + lrow;
        const float* a2p = a2 + (size_t)e * Nn + i0 + half * 8;
#pragma unroll
        for (int m = 0; m < 8; ++m) p[m] += acc[et][m] * a2p[m];
    }
#pragma unroll
    for (int m = 0; m < 8; ++m) {
        p[m] += __shfl_xor(p[m], 1);
        p[m] += __shfl_xor(p[m], 2);
        p[m] += __shfl_xor(p[m], 4);
        p[m] += __shfl_xor(p[m], 8);
    }
    if (lrow == 0) {
        int base = b * Nn + i0 + half * 8;
#pragma unroll
        for (int m = 0; m < 8; ++m) atomicAdd(&score[base + m], p[m]);
    }
}

// ---------------- final: out = leaky_relu(score[b,i] * h_sum[b,e]) ---------------
__global__ __launch_bounds__(256) void final_kernel(const float* __restrict__ score,
                                                    const float* __restrict__ h_sum,
                                                    float* __restrict__ out) {
    size_t idx = ((size_t)blockIdx.x * 256 + threadIdx.x) * 4;
    size_t row = idx >> 8;            // b*N + i
    int b = (int)(row >> 11);
    int e = (int)(idx & 255);
    float s = score[row];
    float4 hs = *reinterpret_cast<const float4*>(h_sum + b * Ee + e);
    float4 v;
    v.x = s * hs.x; v.y = s * hs.y; v.z = s * hs.z; v.w = s * hs.w;
    v.x = v.x > 0.f ? v.x : 0.01f * v.x;
    v.y = v.y > 0.f ? v.y : 0.01f * v.y;
    v.z = v.z > 0.f ? v.z : 0.01f * v.z;
    v.w = v.w > 0.f ? v.w : 0.01f * v.w;
    *reinterpret_cast<float4*>(out + idx) = v;
}

extern "C" void kernel_launch(void* const* d_in, const int* in_sizes, int n_in,
                              void* d_out, int out_size, void* d_ws, size_t ws_size,
                              hipStream_t stream) {
    (void)in_sizes; (void)n_in; (void)out_size; (void)ws_size;
    const float* x   = (const float*)d_in[0];
    const int*   adj = (const int*)d_in[1];
    const float* W_w = (const float*)d_in[2];
    const float* W_b = (const float*)d_in[3];
    const float* a   = (const float*)d_in[4];
    float* out = (float*)d_out;

    // workspace carve-up (all 16B aligned)
    char* ws = (char*)d_ws;
    u16*   maskT = (u16*)ws;                         ws += (size_t)Nn * Nn * 2;        // 8 MB
    u16*   Wt_hi = (u16*)ws;                         ws += (size_t)Ee * Ff * 2;        // 256 KB
    u16*   Wt_lo = (u16*)ws;                         ws += (size_t)Ee * Ff * 2;        // 256 KB
    u16*   ht_hi = (u16*)ws;                         ws += (size_t)Bq * Ee * Nn * 2;   // 32 MB
    u16*   ht_lo = (u16*)ws;                         ws += (size_t)Bq * Ee * Nn * 2;   // 32 MB
    float* score = (float*)ws;                       ws += (size_t)Bq * Nn * 4;        // 256 KB
    float* h_sum = (float*)ws;                       ws += (size_t)Bq * Ee * 4;        // 32 KB

    (void)hipMemsetAsync(score, 0, (size_t)Bq * Nn * 4, stream);
    (void)hipMemsetAsync(h_sum, 0, (size_t)Bq * Ee * 4, stream);

    prepW_kernel<<<(Ee * Ff) / 256, 256, 0, stream>>>(W_w, Wt_hi, Wt_lo);
    prepM_kernel<<<(Nn * Nn) / 256, 256, 0, stream>>>(adj, maskT);

    gemm1_kernel<<<(Bq * Nn) / 64, 256, 0, stream>>>(x, Wt_hi, Wt_lo, W_b, a,
                                                     ht_hi, ht_lo, score, h_sum);
    gemm2_kernel<<<(Bq * Nn) / 64, 256, 0, stream>>>(maskT, ht_hi, ht_lo,
                                                     a + (size_t)Ee * Nn, score);
    final_kernel<<<(Bq * Nn * Ee) / 1024, 256, 0, stream>>>(score, h_sum, out);
}
